// PointTransformer_55009941127297
// MI455X (gfx1250) — compile-verified, matrix operand order checked by reading, not run on hardware
//
#include <hip/hip_runtime.h>

// ---------------------------------------------------------------------------
// Problem constants (match reference)
// ---------------------------------------------------------------------------
#define BB   2
#define NN   4096
#define KNB  20          // neighbors
#define DD   128
#define PHH  8
#define AHH  512
#define BN   (BB*NN)     // 8192 points

typedef __attribute__((ext_vector_type(16))) __bf16 v16bf;
typedef __attribute__((ext_vector_type(8)))  __bf16 v8bf;
typedef __attribute__((ext_vector_type(8)))  float  v8f;

// ---------------------------------------------------------------------------
// WMMA fragment loader: 16x32 bf16 block from a row-major matrix (ld = row
// stride in elements).  Matches CDNA5 16-bit A/B layout:
//   lanes 0-15 : K = 0..7 (vgpr0-3) and 16..23 (vgpr4-7)
//   lanes 16-31: K = 8..15          and 24..31
// For B we exploit that W is stored [N,K] row-major, so X @ W^T needs the
// identical row-major load with lane = n.
// ---------------------------------------------------------------------------
__device__ __forceinline__ v16bf frag_ld(const __bf16* blk, int ld, int lane) {
    int r  = lane & 15;
    int hi = lane >> 4;               // 0 or 1
    const __bf16* p = blk + (size_t)r * ld + hi * 8;
    v8bf c0 = *(const v8bf*)(p);
    v8bf c1 = *(const v8bf*)(p + 16);
    v16bf f;
#pragma unroll
    for (int j = 0; j < 8; ++j) { f[j] = c0[j]; f[j + 8] = c1[j]; }
    return f;
}

// ---------------------------------------------------------------------------
// f32 -> bf16 conversion
// ---------------------------------------------------------------------------
__global__ void __launch_bounds__(256) cvt_bf16_kernel(const float* __restrict__ src,
                                                       __bf16* __restrict__ dst, int n) {
    int i = blockIdx.x * 256 + threadIdx.x;
    if (i < n) dst[i] = (__bf16)src[i];
}

// ---------------------------------------------------------------------------
// linear1: h = x @ W1^T + b1   (K=3, pure VALU), writes f32 + bf16 copies
// ---------------------------------------------------------------------------
__global__ void __launch_bounds__(256) linear1_kernel(const float* __restrict__ x,
                                                      const float* __restrict__ W1,
                                                      const float* __restrict__ b1,
                                                      float* __restrict__ hF,
                                                      __bf16* __restrict__ hBF) {
    int i = blockIdx.x * 256 + threadIdx.x;          // over BN*DD
    if (i >= BN * DD) return;
    int p = i >> 7, d = i & 127;
    const float* xp = x + (size_t)p * 3;
    const float* w  = W1 + d * 3;
    float v = fmaf(xp[0], w[0], fmaf(xp[1], w[1], fmaf(xp[2], w[2], b1[d])));
    hF[i]  = v;
    hBF[i] = (__bf16)v;
}

// ---------------------------------------------------------------------------
// Generic WMMA GEMM: C[M,N] = A[M,Kd] @ W[N,Kd]^T (+bias). One 16x16 tile per
// wave, 8 waves per block. Emits v_wmma_f32_16x16x32_bf16.
// ---------------------------------------------------------------------------
__global__ void __launch_bounds__(256) gemm_bf16_kernel(const __bf16* __restrict__ A,
                                                        const __bf16* __restrict__ W,
                                                        const float* __restrict__ bias,
                                                        float* __restrict__ Cf,
                                                        __bf16* __restrict__ Cbf,
                                                        int M, int N, int Kd) {
    int lane = threadIdx.x & 31;
    int wave = threadIdx.x >> 5;
    int tile = blockIdx.x * 8 + wave;
    int ntiles = N >> 4;
    int mt = tile / ntiles, nt = tile % ntiles;
    if (mt >= (M >> 4)) return;

    v8f acc = {};
    const __bf16* Ab = A + (size_t)(mt * 16) * Kd;
    const __bf16* Wb = W + (size_t)(nt * 16) * Kd;
    for (int ks = 0; ks < Kd; ks += 32) {
        v16bf a = frag_ld(Ab + ks, Kd, lane);
        v16bf b = frag_ld(Wb + ks, Kd, lane);
        acc = __builtin_amdgcn_wmma_f32_16x16x32_bf16(false, a, false, b,
                                                      (short)0, acc, false, false);
    }
    int nn = lane & 15, hi = lane >> 4;
    int col = nt * 16 + nn;
    float bv = bias ? bias[col] : 0.0f;
#pragma unroll
    for (int r = 0; r < 8; ++r) {
        int row = mt * 16 + r + hi * 8;
        float v = acc[r] + bv;
        size_t o = (size_t)row * N + col;
        if (Cf)  Cf[o]  = v;
        if (Cbf) Cbf[o] = (__bf16)v;
    }
}

// ---------------------------------------------------------------------------
// KNN: per-query brute-force top-20 over N candidates (same batch), candidate
// positions tiled through LDS. One thread per query, 128 queries per block.
// ---------------------------------------------------------------------------
__global__ void __launch_bounds__(128) knn_kernel(const float* __restrict__ pos,
                                                  int* __restrict__ idxOut) {
    __shared__ float sP[128][3];
    int tid = threadIdx.x;
    int q   = blockIdx.x * 128 + tid;         // global point id in [0, BN)
    int b   = q / NN;
    int n   = q % NN;
    const float* pb = pos + (size_t)b * NN * 3;
    float qx = pb[n * 3 + 0], qy = pb[n * 3 + 1], qz = pb[n * 3 + 2];

    float bd[KNB]; int bi[KNB];
#pragma unroll
    for (int i = 0; i < KNB; ++i) { bd[i] = 3.4e38f; bi[i] = 0; }

    for (int c = 0; c < NN / 128; ++c) {
        int j0 = c * 128;
        sP[tid][0] = pb[(j0 + tid) * 3 + 0];
        sP[tid][1] = pb[(j0 + tid) * 3 + 1];
        sP[tid][2] = pb[(j0 + tid) * 3 + 2];
        __syncthreads();
        for (int t = 0; t < 128; ++t) {
            float dx = qx - sP[t][0], dy = qy - sP[t][1], dz = qz - sP[t][2];
            float d2 = fmaf(dx, dx, fmaf(dy, dy, dz * dz));
            if (d2 < bd[KNB - 1]) {
                int p = KNB - 1;
                while (p > 0 && bd[p - 1] > d2) {
                    bd[p] = bd[p - 1]; bi[p] = bi[p - 1]; --p;
                }
                bd[p] = d2; bi[p] = j0 + t;
            }
        }
        __syncthreads();
    }
#pragma unroll
    for (int i = 0; i < KNB; ++i) idxOut[(size_t)q * KNB + i] = bi[i];
}

// ---------------------------------------------------------------------------
// Fused attention: one workgroup (256 thr, 8 waves) per query point.
//   async-gather k||v neighbor rows into LDS     (GLOBAL_LOAD_ASYNC_TO_LDS)
//   X = q - k_n + pe  -> [32,128] bf16 (rows 20..31 zero)
//   H = relu(X @ Wa1^T + ba1) -> [32,512] bf16 (WMMA, 8 acc per wave)
//   S = H @ Wa2^T + ba2       -> [20,128] f32  (WMMA, 2 acc per wave)
//   softmax over neighbors per channel, agg = sum attn*(v+pe)
//   out = h + agg @ W3^T + b3
// ---------------------------------------------------------------------------
__global__ void __launch_bounds__(256) attn_kernel(const float* __restrict__ pos,
                                                   const float* __restrict__ hF,
                                                   const float* __restrict__ qkvF,
                                                   const int*   __restrict__ idxB,
                                                   const __bf16* __restrict__ Wa1b,
                                                   const float* __restrict__ ba1,
                                                   const __bf16* __restrict__ Wa2b,
                                                   const float* __restrict__ ba2,
                                                   const float* __restrict__ Wp1,
                                                   const float* __restrict__ bp1,
                                                   const float* __restrict__ Wp2,
                                                   const float* __restrict__ bp2,
                                                   const float* __restrict__ W3,
                                                   const float* __restrict__ b3,
                                                   float* __restrict__ out) {
    __shared__ __align__(16) __bf16 sX[32 * 128];     //  8 KB
    __shared__ __align__(16) __bf16 sH[32 * 512];     // 32 KB
    __shared__ __align__(16) float sKV[KNB * 256];    // 20 KB: k row | v row
    __shared__ float sV[KNB * 132];                   // pe, then v + pe
    __shared__ float sS[KNB * 132];                   // sim
    __shared__ float sQ[128];
    __shared__ float sAgg[128];
    __shared__ int   sIdx[KNB];

    int tid  = threadIdx.x;
    int lane = tid & 31;
    int wave = tid >> 5;
    int qid  = blockIdx.x;            // 0..BN-1
    int b    = qid / NN;
    int n    = qid % NN;

    if (tid < KNB) sIdx[tid] = idxB[(size_t)qid * KNB + tid];
    if (tid < 128) sQ[tid] = qkvF[(size_t)qid * (3 * DD) + tid];

    // keep the 256KB bf16 attn weights hot (global_prefetch_b8)
    {
        const char* w1 = (const char*)Wa1b;
        const char* w2 = (const char*)Wa2b;
        __builtin_prefetch(w1 + tid * 512, 0, 1);
        __builtin_prefetch(w2 + tid * 512, 0, 1);
    }
    __syncthreads();

    // ---- phase 0: async-gather neighbor k||v rows straight into LDS ----
    // neighbor kk handled by wave (kk & 7); each lane moves 16B, two issues
    // cover the 1KB k||v row (256 f32) of that neighbor.
    for (int kk = wave; kk < KNB; kk += 8) {
        int j = sIdx[kk];
        const float* g = qkvF + ((size_t)b * NN + j) * (3 * DD) + DD;   // k row start
        unsigned long long ga = (unsigned long long)(g + lane * 4);
        unsigned int la = (unsigned int)(unsigned long long)(uintptr_t)(sKV + kk * 256 + lane * 4);
        asm volatile("global_load_async_to_lds_b128 %0, %1, off"
                     :: "v"(la), "v"(ga) : "memory");
        unsigned long long ga2 = (unsigned long long)(g + DD + lane * 4);
        unsigned int la2 = (unsigned int)(unsigned long long)(uintptr_t)(sKV + kk * 256 + DD + lane * 4);
        asm volatile("global_load_async_to_lds_b128 %0, %1, off"
                     :: "v"(la2), "v"(ga2) : "memory");
    }

    // ---- phase 1a: pos_mlp -> pe stored in sV (overlaps async gather) ----
    {
        int d    = tid & 127;
        int half = tid >> 7;
        float wp2r[PHH];
#pragma unroll
        for (int e = 0; e < PHH; ++e) wp2r[e] = Wp2[d * PHH + e];
        float bp2d = bp2[d];
        const float* pn = pos + ((size_t)b * NN + n) * 3;
        float p0 = pn[0], p1 = pn[1], p2 = pn[2];

        for (int kk = half; kk < KNB; kk += 2) {
            int j = sIdx[kk];
            const float* pj = pos + ((size_t)b * NN + j) * 3;
            float r0 = p0 - pj[0], r1 = p1 - pj[1], r2 = p2 - pj[2];
            float pe = bp2d;
#pragma unroll
            for (int e = 0; e < PHH; ++e) {
                float hsum = fmaf(r0, Wp1[e * 3 + 0],
                             fmaf(r1, Wp1[e * 3 + 1],
                             fmaf(r2, Wp1[e * 3 + 2], bp1[e])));
                hsum = fmaxf(hsum, 0.0f);
                pe = fmaf(hsum, wp2r[e], pe);
            }
            sV[kk * 132 + d] = pe;
        }
        // zero pad rows 20..31 of X
        for (int t = tid; t < 12 * 128; t += 256)
            sX[(KNB + (t >> 7)) * 128 + (t & 127)] = (__bf16)0.0f;
    }
    asm volatile("s_wait_asynccnt 0" ::: "memory");
    __syncthreads();

    // ---- phase 1b: combine gathered k/v with pe ----
    {
        int d    = tid & 127;
        int half = tid >> 7;
        float qd = sQ[d];
        for (int kk = half; kk < KNB; kk += 2) {
            float pe = sV[kk * 132 + d];
            float kn = sKV[kk * 256 + d];
            float vn = sKV[kk * 256 + DD + d];
            sX[kk * 128 + d] = (__bf16)(qd - kn + pe);
            sV[kk * 132 + d] = vn + pe;
        }
    }
    __syncthreads();

    // ---- phase 2: H = relu(X @ Wa1^T + ba1), bf16 into LDS ----
    // wave -> mt = wave&1, nt = (wave>>1) + 4*i  (8 acc, A-frag reuse x8)
    {
        int mt  = wave & 1;
        int ntb = wave >> 1;              // 0..3
        v8f acc[8];
#pragma unroll
        for (int i = 0; i < 8; ++i) acc[i] = (v8f){};
        const __bf16* Ab = sX + (mt * 16) * 128;
#pragma unroll
        for (int ks = 0; ks < 128; ks += 32) {
            v16bf a = frag_ld(Ab + ks, 128, lane);
#pragma unroll
            for (int i = 0; i < 8; ++i) {
                int nt = ntb + i * 4;
                v16bf bb = frag_ld(Wa1b + (size_t)(nt * 16) * 128 + ks, 128, lane);
                acc[i] = __builtin_amdgcn_wmma_f32_16x16x32_bf16(false, a, false, bb,
                                                                 (short)0, acc[i], false, false);
            }
        }
        int nn = lane & 15, hi = lane >> 4;
#pragma unroll
        for (int i = 0; i < 8; ++i) {
            int col = (ntb + i * 4) * 16 + nn;
            float bv = ba1[col];
#pragma unroll
            for (int r = 0; r < 8; ++r) {
                int row = mt * 16 + r + hi * 8;
                sH[row * 512 + col] = (__bf16)fmaxf(acc[i][r] + bv, 0.0f);
            }
        }
    }
    __syncthreads();

    // ---- phase 3: S = H @ Wa2^T + ba2 (rows < 20 kept) ----
    // wave -> mt = wave&1, nt = (wave>>1) and (wave>>1)+4  (A-frag reuse x2)
    {
        int mt  = wave & 1;
        int nt0 = wave >> 1;              // 0..3
        v8f acc0 = {}, acc1 = {};
        const __bf16* Ab = sH + (mt * 16) * 512;
        const __bf16* Wb0 = Wa2b + (size_t)(nt0 * 16) * 512;
        const __bf16* Wb1 = Wa2b + (size_t)((nt0 + 4) * 16) * 512;
        for (int ks = 0; ks < 512; ks += 32) {
            v16bf a  = frag_ld(Ab + ks, 512, lane);
            v16bf b0 = frag_ld(Wb0 + ks, 512, lane);
            v16bf b1 = frag_ld(Wb1 + ks, 512, lane);
            acc0 = __builtin_amdgcn_wmma_f32_16x16x32_bf16(false, a, false, b0,
                                                           (short)0, acc0, false, false);
            acc1 = __builtin_amdgcn_wmma_f32_16x16x32_bf16(false, a, false, b1,
                                                           (short)0, acc1, false, false);
        }
        int nn = lane & 15, hi = lane >> 4;
#pragma unroll
        for (int r = 0; r < 8; ++r) {
            int row = mt * 16 + r + hi * 8;
            if (row < KNB) {
                int c0 = nt0 * 16 + nn;
                int c1 = (nt0 + 4) * 16 + nn;
                sS[row * 132 + c0] = acc0[r] + ba2[c0];
                sS[row * 132 + c1] = acc1[r] + ba2[c1];
            }
        }
    }
    __syncthreads();

    // ---- phase 4: softmax over the 20 neighbors, weighted sum ----
    if (tid < 128) {
        int d = tid;
        float mx = -3.4e38f;
#pragma unroll
        for (int kk = 0; kk < KNB; ++kk) mx = fmaxf(mx, sS[kk * 132 + d]);
        float den = 0.0f, agg = 0.0f;
#pragma unroll
        for (int kk = 0; kk < KNB; ++kk) {
            float e = __expf(sS[kk * 132 + d] - mx);
            den += e;
            agg = fmaf(e, sV[kk * 132 + d], agg);
        }
        sAgg[d] = agg / den;
    }
    __syncthreads();

    // ---- phase 5: out = h + agg @ W3^T + b3 ----
    if (tid < 128) {
        int d = tid;
        const float* w = W3 + d * DD;
        float acc = b3[d];
#pragma unroll 8
        for (int c = 0; c < DD; ++c) acc = fmaf(sAgg[c], w[c], acc);
        out[(size_t)qid * DD + d] = hF[(size_t)qid * DD + d] + acc;
    }
}

// ---------------------------------------------------------------------------
// Host launcher
// ---------------------------------------------------------------------------
extern "C" void kernel_launch(void* const* d_in, const int* in_sizes, int n_in,
                              void* d_out, int out_size, void* d_ws, size_t ws_size,
                              hipStream_t stream) {
    const float* x    = (const float*)d_in[0];
    const float* pos  = (const float*)d_in[1];
    const float* W1   = (const float*)d_in[2];
    const float* b1   = (const float*)d_in[3];
    const float* W2   = (const float*)d_in[4];
    const float* b2   = (const float*)d_in[5];
    const float* W3   = (const float*)d_in[6];
    const float* b3   = (const float*)d_in[7];
    const float* Wqkv = (const float*)d_in[8];
    const float* Wp1  = (const float*)d_in[9];
    const float* bp1  = (const float*)d_in[10];
    const float* Wp2  = (const float*)d_in[11];
    const float* bp2  = (const float*)d_in[12];
    const float* Wa1  = (const float*)d_in[13];
    const float* ba1  = (const float*)d_in[14];
    const float* Wa2  = (const float*)d_in[15];
    const float* ba2  = (const float*)d_in[16];
    float* out = (float*)d_out;

    // workspace carve (256B aligned)
    char* ws = (char*)d_ws;
    size_t off = 0;
    auto alloc = [&](size_t bytes) -> void* {
        void* p = ws + off;
        off = (off + bytes + 255) & ~(size_t)255;
        return p;
    };
    float*  hF     = (float*) alloc((size_t)BN * DD * 4);
    __bf16* hBF    = (__bf16*)alloc((size_t)BN * DD * 2);
    __bf16* h1BF   = (__bf16*)alloc((size_t)BN * DD * 2);
    float*  qkvF   = (float*) alloc((size_t)BN * 3 * DD * 4);
    int*    idxB   = (int*)   alloc((size_t)BN * KNB * 4);
    __bf16* W2b    = (__bf16*)alloc((size_t)DD * DD * 2);
    __bf16* Wqkvb  = (__bf16*)alloc((size_t)3 * DD * DD * 2);
    __bf16* Wa1b   = (__bf16*)alloc((size_t)AHH * DD * 2);
    __bf16* Wa2b   = (__bf16*)alloc((size_t)DD * AHH * 2);
    (void)ws_size; (void)in_sizes; (void)n_in; (void)out_size;

    // 1. weight downconverts
    cvt_bf16_kernel<<<(DD * DD + 255) / 256, 256, 0, stream>>>(W2, W2b, DD * DD);
    cvt_bf16_kernel<<<(3 * DD * DD + 255) / 256, 256, 0, stream>>>(Wqkv, Wqkvb, 3 * DD * DD);
    cvt_bf16_kernel<<<(AHH * DD + 255) / 256, 256, 0, stream>>>(Wa1, Wa1b, AHH * DD);
    cvt_bf16_kernel<<<(DD * AHH + 255) / 256, 256, 0, stream>>>(Wa2, Wa2b, DD * AHH);

    // 2. linear1
    linear1_kernel<<<(BN * DD + 255) / 256, 256, 0, stream>>>(x, W1, b1, hF, hBF);

    // 3. h1 = h @ W2^T + b2   (M=8192,N=128,K=128) -> bf16
    {
        int tiles = (BN / 16) * (DD / 16);
        gemm_bf16_kernel<<<tiles / 8, 256, 0, stream>>>(hBF, W2b, b2, nullptr, h1BF,
                                                        BN, DD, DD);
    }
    // 4. qkv = h1 @ Wqkv^T    (M=8192,N=384,K=128) -> f32
    {
        int tiles = (BN / 16) * (3 * DD / 16);
        gemm_bf16_kernel<<<tiles / 8, 256, 0, stream>>>(h1BF, Wqkvb, nullptr, qkvF, nullptr,
                                                        BN, 3 * DD, DD);
    }
    // 5. KNN
    knn_kernel<<<BN / 128, 128, 0, stream>>>(pos, idxB);

    // 6. fused attention + linear3 + residual
    attn_kernel<<<BN, 256, 0, stream>>>(pos, hF, qkvF, idxB,
                                        Wa1b, ba1, Wa2b, ba2,
                                        Wp1, bp1, Wp2, bp2,
                                        W3, b3, out);
}
